// BilateralUpsampleNet_32229434589813
// MI455X (gfx1250) — compile-verified, matrix operand order checked by reading, not run on hardware
//
#include <hip/hip_runtime.h>

typedef __attribute__((ext_vector_type(16))) _Float16 v16h;
typedef __attribute__((ext_vector_type(8)))  float    v8f;
typedef __attribute__((ext_vector_type(4)))  unsigned int v4u;
typedef __attribute__((ext_vector_type(8)))  int      v8i_;
typedef __attribute__((ext_vector_type(4)))  int      v4i_;

#define NIMG 2
#define HH 1080
#define WW 1920
#define ROWS_PER_WG 4
#define TILES_X (WW / 32)                      // 60 tiles of 32 px per row
#define TILES_PER_WG (ROWS_PER_WG * TILES_X)   // 240
#define NWAVES 8

// Fused HDRNet: TDM grid fill -> guide MLP (WMMA) -> LDS trilerp -> affine apply.
__global__ __launch_bounds__(256) void hdrnet_fused(
    const float* __restrict__ fr,  const float* __restrict__ bg,
    const float* __restrict__ w1,  const float* __restrict__ b1,
    const float* __restrict__ bng, const float* __restrict__ bnb,
    const float* __restrict__ bnm, const float* __restrict__ bnv,
    const float* __restrict__ w2,  const float* __restrict__ b2,
    float* __restrict__ out)
{
  // 3 grid rows cover a 4-pixel-row strip: 12ch x 8z x 3y x 16x f32 = 18KB.
  // LDS layout (TDM-compacted): ((cz)*3 + j)*16 + x, cz = c*8+z (stride-256 slices).
  __shared__ float    s_grid[96 * 3 * 16];
  __shared__ _Float16 s_relu[NWAVES][32][16];  // conv1 activations (transposed)
  __shared__ _Float16 s_ast [NWAVES][32][4];   // per-pixel {r,g,b,1} f16 staging

  const int tid  = threadIdx.x;
  const int wv   = tid >> 5;
  const int lane = tid & 31;
  const int blk  = blockIdx.x;
  const int n    = blk / (HH / ROWS_PER_WG);
  const int rb   = blk % (HH / ROWS_PER_WG);
  const int ybase = rb * ROWS_PER_WG;

  // Which grid rows does this strip touch? floor(gy) varies by at most 1
  // across 4 pixel rows, and y1 = y0+1 -> 3 cached rows suffice.
  float gyb = ybase * (15.0f / 1079.0f);
  int ry0 = (int)floorf(gyb);
  if (ry0 > 14) ry0 = 14;

  // ---- Async TDM fill of the grid slab (overlapped with weight prep) ----
  // Tile: dim0 = 16 x-texels (stride 1), dim1 = 3 rows (stride 16),
  //       dim2 = 96 (c,z)-slices (stride 256). Row overhang past row 15
  //       zero-fills via tensor_dim1 = remaining rows (that slot is never read).
  const float* gsrc = bg + (size_t)n * (12 * 8 * 16 * 16);
  if (wv == 0) {
    const unsigned long long gaddr =
        (unsigned long long)(const void*)(gsrc + ry0 * 16);
    // Generic LDS VAs are {aperture_hi : lds_offset}; low 32 bits = LDS address.
    const unsigned lds_base = (unsigned)(unsigned long long)(void*)&s_grid[0];

    v4u g0;
    g0[0] = 1u;                                        // count=1, no gather
    g0[1] = lds_base;                                  // lds_addr
    g0[2] = (unsigned)(gaddr & 0xFFFFFFFFu);           // global_addr[31:0]
    g0[3] = (unsigned)((gaddr >> 32) & 0x01FFFFFFu)    // global_addr[56:32]
          | (2u << 30);                                // type = 2 (image)
    v8i_ g1;
    g1[0] = (2 << 16);                                 // data_size = 4B
    g1[1] = (16 << 16);                                // tensor_dim0 = 16
    g1[2] = ((16 - ry0) << 16);                        // tensor_dim1 = rows left
    g1[3] = (16 << 16);                                // tile_dim0 = 16
    g1[4] = 3 | (96 << 16);                            // tile_dim1=3, tile_dim2=96
    g1[5] = 16;                                        // tensor_dim0_stride = 16
    g1[6] = (256 << 16);                               // tensor_dim1_stride = 256
    g1[7] = 0;
    v4i_ g2;
    g2[0] = 96;                                        // tensor_dim2
    g2[1] = 0; g2[2] = 0; g2[3] = 0;
    v4i_ g3;
    g3[0] = 0; g3[1] = 0; g3[2] = 0; g3[3] = 0;
    v8i_ g4;                                           // unused 5th group (VADDR4)
    #pragma unroll
    for (int i = 0; i < 8; ++i) g4[i] = 0;
    __builtin_amdgcn_tensor_load_to_lds(g0, g1, g2, g3, g4, 0);
  }

  // Fold BN into conv1 while the DMA is in flight.
  // B matrix (K=32 x N=16 f16): lanes 0..15 hold K=0..15 for column N=lane.
  const int ch = lane & 15;
  v16h bmat;
  #pragma unroll
  for (int i = 0; i < 16; ++i) bmat[i] = (_Float16)0.0f;
  if (lane < 16) {
    float inv = bng[ch] * rsqrtf(bnv[ch] + 1e-5f);
    bmat[0] = (_Float16)(w1[ch * 3 + 0] * inv);
    bmat[1] = (_Float16)(w1[ch * 3 + 1] * inv);
    bmat[2] = (_Float16)(w1[ch * 3 + 2] * inv);
    bmat[3] = (_Float16)((b1[ch] - bnm[ch]) * inv + bnb[ch]);
  }
  float w2r[16];
  #pragma unroll
  for (int i = 0; i < 16; ++i) w2r[i] = w2[i];   // uniform -> scalar loads
  const float b2v = b2[0];

  v8f cz;
  #pragma unroll
  for (int i = 0; i < 8; ++i) cz[i] = 0.0f;

  if (wv == 0) __builtin_amdgcn_s_wait_tensorcnt(0);
  __syncthreads();

  const size_t plane = (size_t)HH * WW;
  const float* frn = fr  + (size_t)n * 3 * plane;
  float*      outn = out + (size_t)n * 3 * plane;

  for (int t = wv; t < TILES_PER_WG; t += NWAVES) {
    const int yrow = ybase + t / TILES_X;
    const int xb   = (t % TILES_X) * 32;
    const int x    = xb + lane;
    const size_t pix = (size_t)yrow * WW + x;

    // Prefetch next tile's red plane (gfx1250 global_prefetch_b8).
    if (t + NWAVES < TILES_PER_WG) {
      int t2 = t + NWAVES;
      size_t p2 = (size_t)(ybase + t2 / TILES_X) * WW + (t2 % TILES_X) * 32 + lane;
      __builtin_prefetch(frn + p2, 0, 0);
    }

    const float rr = frn[pix];
    const float gg = frn[plane + pix];
    const float bb = frn[2 * plane + pix];

    // Stage {r,g,b,1} as f16 so lanes 0..15 can gather A rows for both tiles.
    s_ast[wv][lane][0] = (_Float16)rr;
    s_ast[wv][lane][1] = (_Float16)gg;
    s_ast[wv][lane][2] = (_Float16)bb;
    s_ast[wv][lane][3] = (_Float16)1.0f;

    // A layout (f16 16x32): lanes 0..15 carry K=0..7 (h[0..3] used), rest zero.
    v16h a0, a1;
    #pragma unroll
    for (int i = 0; i < 16; ++i) { a0[i] = (_Float16)0.0f; a1[i] = (_Float16)0.0f; }
    if (lane < 16) {
      #pragma unroll
      for (int i = 0; i < 4; ++i) {
        a0[i] = s_ast[wv][lane][i];        // pixels 0..15 of tile
        a1[i] = s_ast[wv][lane + 16][i];   // pixels 16..31 of tile
      }
    }

    // conv1 (+bias, BN folded) for 32 pixels in two WMMAs.
    v8f c0 = __builtin_amdgcn_wmma_f32_16x16x32_f16(false, a0, false, bmat,
                                                    (short)0, cz, false, false);
    v8f c1 = __builtin_amdgcn_wmma_f32_16x16x32_f16(false, a1, false, bmat,
                                                    (short)0, cz, false, false);

    // D layout: lane l, vgpr r -> pixel (r + 8*(l>=16)), channel l%16.
    const int hl = lane >> 4;
    #pragma unroll
    for (int r = 0; r < 8; ++r) {
      s_relu[wv][ 0 + r + 8 * hl][ch] = (_Float16)fmaxf(c0[r], 0.0f);
      s_relu[wv][16 + r + 8 * hl][ch] = (_Float16)fmaxf(c1[r], 0.0f);
    }

    // conv2: each lane dots its own pixel's 16 channels; sigmoid.
    float acc = b2v;
    #pragma unroll
    for (int i = 0; i < 16; ++i) acc += (float)s_relu[wv][lane][i] * w2r[i];
    const float guide = 1.0f / (1.0f + __expf(-acc));

    // ---- trilinear slice of the cached grid slab ----
    float gz = guide * 7.0f;
    int z0 = (int)floorf(gz);
    float tz = gz - (float)z0;
    if (z0 < 0) { z0 = 0; tz = 0.0f; }
    if (z0 > 6) { z0 = 6; tz = 1.0f; }

    float gyl = yrow * (15.0f / 1079.0f);
    int y0 = (int)floorf(gyl);
    float ty = gyl - (float)y0;
    if (y0 > 14) { y0 = 14; ty = 1.0f; }
    const int yl = y0 - ry0;               // 0 or 1 inside the slab

    float gx = x * (15.0f / 1919.0f);
    int x0 = (int)floorf(gx);
    float tx = gx - (float)x0;
    if (x0 > 14) { x0 = 14; tx = 1.0f; }

    // slab layout: ((c*8 + z)*3 + j)*16 + x
    const float* qb = s_grid + (z0 * 3 + yl) * 16 + x0;
    float coeff[12];
    #pragma unroll
    for (int c = 0; c < 12; ++c) {
      const float* q = qb + c * (8 * 3 * 16);
      float a00 = q[0],  a01 = q[1],  a10 = q[16], a11 = q[17];   // z0
      float b00 = q[48], b01 = q[49], b10 = q[64], b11 = q[65];   // z1
      float xa  = a00 + tx * (a01 - a00);
      float xb2 = a10 + tx * (a11 - a10);
      float xc  = b00 + tx * (b01 - b00);
      float xd  = b10 + tx * (b11 - b10);
      float ya  = xa + ty * (xb2 - xa);
      float yb  = xc + ty * (xd - xc);
      coeff[c] = ya + tz * (yb - ya);
    }

    // ---- affine apply ----
    outn[pix]             = coeff[0] * rr + coeff[1] * gg + coeff[2]  * bb + coeff[3];
    outn[plane + pix]     = coeff[4] * rr + coeff[5] * gg + coeff[6]  * bb + coeff[7];
    outn[2 * plane + pix] = coeff[8] * rr + coeff[9] * gg + coeff[10] * bb + coeff[11];
  }
}

extern "C" void kernel_launch(void* const* d_in, const int* in_sizes, int n_in,
                              void* d_out, int out_size, void* d_ws, size_t ws_size,
                              hipStream_t stream) {
  (void)in_sizes; (void)n_in; (void)out_size; (void)d_ws; (void)ws_size;
  const float* fr  = (const float*)d_in[0];
  const float* bg  = (const float*)d_in[1];
  const float* w1  = (const float*)d_in[2];
  const float* b1  = (const float*)d_in[3];
  const float* bng = (const float*)d_in[4];
  const float* bnb = (const float*)d_in[5];
  const float* bnm = (const float*)d_in[6];
  const float* bnv = (const float*)d_in[7];
  const float* w2  = (const float*)d_in[8];
  const float* b2  = (const float*)d_in[9];
  float* out = (float*)d_out;

  dim3 grid(NIMG * (HH / ROWS_PER_WG));   // 540 workgroups
  dim3 block(256);                        // 8 waves (wave32)
  hdrnet_fused<<<grid, block, 0, stream>>>(fr, bg, w1, b1, bng, bnb, bnm, bnv,
                                           w2, b2, out);
}
